// OrientedRCNNHead_1906965480114
// MI455X (gfx1250) — matrix operand order repaired and checked by reading, not compiled, not run on hardware
//
#include <hip/hip_runtime.h>
#include <hip/hip_bf16.h>

// ---------------------------------------------------------------------------
// OrientedRCNNHead on MI455X (gfx1250, wave32, WMMA).
//
// bf16 WMMA (16x16x32, f32 accumulate): 111 GFLOP vs ~260 MB HBM traffic.
// B tiles: global -> (regs) -> double-buffered LDS (one copy per WG, 8x less
// L2->WGP traffic), one barrier per 4 K-steps. Inside a phase, B fragments
// are additionally double-buffered in REGISTERS one K-step ahead, so the 4
// independent WMMAs of each step issue back-to-back without waiting dscnt 0
// on just-issued ds_loads. A stream is software-pipelined one K-step ahead
// with in-register f32->bf16 conversion.
// ---------------------------------------------------------------------------

typedef __attribute__((ext_vector_type(16))) __bf16 bf16x16;
typedef __attribute__((ext_vector_type(8)))  __bf16 bf16x8;
typedef __attribute__((ext_vector_type(8)))  float  f32x8;
typedef __attribute__((ext_vector_type(8)))  float  v8f;
typedef __attribute__((ext_vector_type(4)))  unsigned int u32x4;

#define M_ROWS 4000      // B*N
#define K1     12544     // C_IN*7*7
#define D1     1024
#define NCLS   11        // num_classes + 1
#define NREG   5

// ---------------------------------------------------------------------------
// Pack f32 [K,N] weights into bf16 WMMA B-fragments:
// layout [kt][nt][lane][j] with lane = ((k>>3)&1)*16 + (n&15), j = (k&7) + ((k>>4)&1)*8.
// Consecutive nt tiles are contiguous, so 4 N-tiles of one K-step form one
// contiguous 4 KB block (perfect for cooperative LDS staging).
// ---------------------------------------------------------------------------
__global__ __launch_bounds__(256) void pack_w_bf16(
    const float* __restrict__ w, __bf16* __restrict__ out, int K, int N)
{
    int idx = blockIdx.x * 256 + threadIdx.x;
    if (idx >= K * N) return;
    int k = idx / N, n = idx - k * N;
    int Nt = N >> 4;
    int kt = k >> 5;
    int j  = (k & 7) + (((k >> 4) & 1) << 3);
    int hi = (k >> 3) & 1;
    int lane = hi * 16 + (n & 15);
    int nt = n >> 4;
    size_t o = (((size_t)(kt * Nt + nt)) * 32 + (size_t)lane) * 16 + (size_t)j;
    out[o] = (__bf16)w[(size_t)k * N + n];
}

// ---------------------------------------------------------------------------
// GEMM1: C[M,1024] = relu(A_f32[M,12544] * Wp_bf16 + bias), output bf16.
// 256 threads = 8 waves; wave w owns rows m0 = bx*128 + w*16, cols n0=by*64.
// ---------------------------------------------------------------------------
__global__ __launch_bounds__(256) void gemm1_wmma(
    const float* __restrict__ A, const __bf16* __restrict__ Bp,
    const float* __restrict__ bias, __bf16* __restrict__ Out,
    int M, int K, int N)
{
    __shared__ __align__(16) unsigned char smem[2][4][4096];  // 32 KB
    const int tid  = threadIdx.x;
    const int lane = tid & 31;
    const int wave = tid >> 5;
    const int m0 = blockIdx.x * 128 + wave * 16;
    const int n0 = blockIdx.y * 64;
    const int Nt  = N >> 4;
    const int nt0 = n0 >> 4;

    int mA = m0 + (lane & 15);
    if (mA > M - 1) mA = M - 1;               // clamp loads, guard stores later
    const int kHi = (lane >> 4) * 8;          // lane<16: K {0..7,16..23}; else {8..15,24..31}
    const float* arow = A + (size_t)mA * K + kHi;

    const int KT = K >> 5;                    // 32-wide K steps
    const int PH = KT >> 2;                   // phases of 4 K-steps

    // Prologue: stage phase 0 B tiles into LDS buffer 0.
#pragma unroll
    for (int s = 0; s < 4; ++s) {
        const __bf16* src = Bp + (size_t)(s * Nt + nt0) * 512 + tid * 8;
        *(u32x4*)(&smem[0][s][tid * 16]) = *(const u32x4*)src;
    }

    v8f acc[4];
#pragma unroll
    for (int i = 0; i < 4; ++i) acc[i] = (v8f){};

    // A pipeline: current K-step in (alo,ahi).
    f32x8 alo = *(const f32x8*)(arow);
    f32x8 ahi = *(const f32x8*)(arow + 16);

    __syncthreads();

    for (int p = 0; p < PH; ++p) {
        // Issue next phase's B global loads now; ds_store them after compute.
        u32x4 stg[4];
        const bool more = (p + 1 < PH);
        if (more) {
#pragma unroll
            for (int s = 0; s < 4; ++s) {
                const __bf16* src =
                    Bp + (size_t)(((p + 1) * 4 + s) * Nt + nt0) * 512 + tid * 8;
                stg[s] = *(const u32x4*)src;
            }
        }
        const unsigned char* buf = &smem[p & 1][0][0];

        // B-fragment register pipeline: preload K-step 0 of this phase.
        bf16x16 bcur[4];
#pragma unroll
        for (int i = 0; i < 4; ++i)
            bcur[i] = *(const bf16x16*)(buf + i * 1024 + lane * 32);

#pragma unroll
        for (int s = 0; s < 4; ++s) {
            bf16x16 bnxt[4];
            if (s < 3) {
#pragma unroll
                for (int i = 0; i < 4; ++i)
                    bnxt[i] = *(const bf16x16*)(buf + (s + 1) * 4096 + i * 1024 + lane * 32);
            }
            const int kt  = p * 4 + s;
            const int ktn = (kt + 1 < KT) ? kt + 1 : kt;
            f32x8 nlo = *(const f32x8*)(arow + (size_t)ktn * 32);
            f32x8 nhi = *(const f32x8*)(arow + (size_t)ktn * 32 + 16);
            bf16x16 a;
#pragma unroll
            for (int e = 0; e < 8; ++e) {
                a[e]     = (__bf16)alo[e];
                a[8 + e] = (__bf16)ahi[e];
            }
#pragma unroll
            for (int i = 0; i < 4; ++i)
                acc[i] = __builtin_amdgcn_wmma_f32_16x16x32_bf16(
                    false, a, false, bcur[i], (short)0, acc[i], false, false);
            alo = nlo; ahi = nhi;
            if (s < 3) {
#pragma unroll
                for (int i = 0; i < 4; ++i) bcur[i] = bnxt[i];
            }
        }
        if (more) {
#pragma unroll
            for (int s = 0; s < 4; ++s)
                *(u32x4*)(&smem[(p + 1) & 1][s][tid * 16]) = stg[s];
        }
        __syncthreads();
    }

    // Epilogue: C layout = VGPR v, lanes 0-15 -> M=v, lanes 16-31 -> M=v+8.
    const int mbase = m0 + ((lane >> 4) * 8);
#pragma unroll
    for (int i = 0; i < 4; ++i) {
        const int n = n0 + i * 16 + (lane & 15);
        const float bv = bias[n];
#pragma unroll
        for (int v = 0; v < 8; ++v) {
            const int m = mbase + v;
            if (m < M) {
                float val = acc[i][v] + bv;
                val = val > 0.f ? val : 0.f;
                Out[(size_t)m * N + n] = (__bf16)val;
            }
        }
    }
}

// ---------------------------------------------------------------------------
// GEMM2: C[M,1024] = relu(A_bf16[M,1024] * Wp_bf16 + bias), output f32.
// ---------------------------------------------------------------------------
__global__ __launch_bounds__(256) void gemm2_wmma(
    const __bf16* __restrict__ A, const __bf16* __restrict__ Bp,
    const float* __restrict__ bias, float* __restrict__ Out,
    int M, int K, int N)
{
    __shared__ __align__(16) unsigned char smem[2][4][4096];  // 32 KB
    const int tid  = threadIdx.x;
    const int lane = tid & 31;
    const int wave = tid >> 5;
    const int m0 = blockIdx.x * 128 + wave * 16;
    const int n0 = blockIdx.y * 64;
    const int Nt  = N >> 4;
    const int nt0 = n0 >> 4;

    int mA = m0 + (lane & 15);
    if (mA > M - 1) mA = M - 1;
    const int kHi = (lane >> 4) * 8;
    const __bf16* arow = A + (size_t)mA * K + kHi;

    const int KT = K >> 5;
    const int PH = KT >> 2;

#pragma unroll
    for (int s = 0; s < 4; ++s) {
        const __bf16* src = Bp + (size_t)(s * Nt + nt0) * 512 + tid * 8;
        *(u32x4*)(&smem[0][s][tid * 16]) = *(const u32x4*)src;
    }

    v8f acc[4];
#pragma unroll
    for (int i = 0; i < 4; ++i) acc[i] = (v8f){};

    bf16x8 alo = *(const bf16x8*)(arow);
    bf16x8 ahi = *(const bf16x8*)(arow + 16);

    __syncthreads();

    for (int p = 0; p < PH; ++p) {
        u32x4 stg[4];
        const bool more = (p + 1 < PH);
        if (more) {
#pragma unroll
            for (int s = 0; s < 4; ++s) {
                const __bf16* src =
                    Bp + (size_t)(((p + 1) * 4 + s) * Nt + nt0) * 512 + tid * 8;
                stg[s] = *(const u32x4*)src;
            }
        }
        const unsigned char* buf = &smem[p & 1][0][0];

        bf16x16 bcur[4];
#pragma unroll
        for (int i = 0; i < 4; ++i)
            bcur[i] = *(const bf16x16*)(buf + i * 1024 + lane * 32);

#pragma unroll
        for (int s = 0; s < 4; ++s) {
            bf16x16 bnxt[4];
            if (s < 3) {
#pragma unroll
                for (int i = 0; i < 4; ++i)
                    bnxt[i] = *(const bf16x16*)(buf + (s + 1) * 4096 + i * 1024 + lane * 32);
            }
            const int kt  = p * 4 + s;
            const int ktn = (kt + 1 < KT) ? kt + 1 : kt;
            bf16x8 nlo = *(const bf16x8*)(arow + (size_t)ktn * 32);
            bf16x8 nhi = *(const bf16x8*)(arow + (size_t)ktn * 32 + 16);
            bf16x16 a;
#pragma unroll
            for (int e = 0; e < 8; ++e) {
                a[e]     = alo[e];
                a[8 + e] = ahi[e];
            }
#pragma unroll
            for (int i = 0; i < 4; ++i)
                acc[i] = __builtin_amdgcn_wmma_f32_16x16x32_bf16(
                    false, a, false, bcur[i], (short)0, acc[i], false, false);
            alo = nlo; ahi = nhi;
            if (s < 3) {
#pragma unroll
                for (int i = 0; i < 4; ++i) bcur[i] = bnxt[i];
            }
        }
        if (more) {
#pragma unroll
            for (int s = 0; s < 4; ++s)
                *(u32x4*)(&smem[(p + 1) & 1][s][tid * 16]) = stg[s];
        }
        __syncthreads();
    }

    const int mbase = m0 + ((lane >> 4) * 8);
#pragma unroll
    for (int i = 0; i < 4; ++i) {
        const int n = n0 + i * 16 + (lane & 15);
        const float bv = bias[n];
#pragma unroll
        for (int v = 0; v < 8; ++v) {
            const int m = mbase + v;
            if (m < M) {
                float val = acc[i][v] + bv;
                Out[(size_t)m * N + n] = val > 0.f ? val : 0.f;
            }
        }
    }
}

// ---------------------------------------------------------------------------
// Head: per row, 11-class scores + 5 regression deltas (K=1024 dots, f32),
// plus boxes_out = boxes + reg and rois = boxes. 16 worker threads per row.
// d_out layout: cls [4000*11] | boxes_out [4000*5] | rois [4000*5].
// ---------------------------------------------------------------------------
__global__ __launch_bounds__(256) void head_kernel(
    const float* __restrict__ h2, const float* __restrict__ boxes,
    const float* __restrict__ wc, const float* __restrict__ bc,
    const float* __restrict__ wr, const float* __restrict__ br,
    float* __restrict__ out, int M)
{
    int t = blockIdx.x * 256 + threadIdx.x;
    int row = t >> 4;
    int o = t & 15;
    if (row >= M) return;
    const float* h = h2 + (size_t)row * D1;
    float s = 0.f;
    if (o < NCLS) {
        for (int d = 0; d < D1; ++d) s += h[d] * wc[d * NCLS + o];
        out[(size_t)row * NCLS + o] = s + bc[o];
    } else {
        int r = o - NCLS;
        for (int d = 0; d < D1; ++d) s += h[d] * wr[d * NREG + r];
        float b = boxes[(size_t)row * NREG + r];
        out[(size_t)M * NCLS + (size_t)row * NREG + r] = b + s + br[r];        // boxes_out
        out[(size_t)M * (NCLS + NREG) + (size_t)row * NREG + r] = b;           // rois
    }
}

// ---------------------------------------------------------------------------
extern "C" void kernel_launch(void* const* d_in, const int* in_sizes, int n_in,
                              void* d_out, int out_size, void* d_ws, size_t ws_size,
                              hipStream_t stream) {
    const float* features = (const float*)d_in[0];  // [2,2000,256,7,7]
    const float* boxes    = (const float*)d_in[1];  // [2,2000,5]
    const float* w1       = (const float*)d_in[2];  // [12544,1024]
    const float* b1       = (const float*)d_in[3];
    const float* w2       = (const float*)d_in[4];  // [1024,1024]
    const float* b2       = (const float*)d_in[5];
    const float* wc       = (const float*)d_in[6];  // [1024,11]
    const float* bc       = (const float*)d_in[7];
    const float* wr       = (const float*)d_in[8];  // [1024,5]
    const float* br       = (const float*)d_in[9];
    float* out = (float*)d_out;

    // Workspace layout (all offsets 32B-aligned for bf16x16 vector loads).
    char* ws = (char*)d_ws;
    const size_t W1P_BYTES = (size_t)K1 * D1 * 2;      // 25,690,112
    const size_t W2P_BYTES = (size_t)D1 * D1 * 2;      //  2,097,152
    const size_t H1_BYTES  = (size_t)M_ROWS * D1 * 2;  //  8,192,000
    __bf16* w1p = (__bf16*)(ws);
    __bf16* w2p = (__bf16*)(ws + W1P_BYTES);
    __bf16* h1  = (__bf16*)(ws + W1P_BYTES + W2P_BYTES);
    float*  h2  = (float*)(ws + W1P_BYTES + W2P_BYTES + H1_BYTES);
    // total ~52.4 MB of d_ws

    // 1) Pack weights into bf16 B-fragments (w1 stream read once; fits in L2).
    {
        int n1 = K1 * D1;
        pack_w_bf16<<<(n1 + 255) / 256, 256, 0, stream>>>(w1, w1p, K1, D1);
        int n2 = D1 * D1;
        pack_w_bf16<<<(n2 + 255) / 256, 256, 0, stream>>>(w2, w2p, D1, D1);
    }

    // 2) h1 = relu(features * w1 + b1)   [dominant GEMM: 102.8 GFLOP]
    {
        dim3 grid((M_ROWS + 127) / 128, D1 / 64);   // 32 x 16 workgroups
        gemm1_wmma<<<grid, 256, 0, stream>>>(features, w1p, b1, h1, M_ROWS, K1, D1);
    }

    // 3) h2 = relu(h1 * w2 + b2)
    {
        dim3 grid((M_ROWS + 127) / 128, D1 / 64);
        gemm2_wmma<<<grid, 256, 0, stream>>>(h1, w2p, b2, h2, M_ROWS, D1, D1);
    }

    // 4) Heads + box add + rois copy.
    {
        int threads = M_ROWS * 16;
        head_kernel<<<(threads + 255) / 256, 256, 0, stream>>>(
            h2, boxes, wc, bc, wr, br, out, M_ROWS);
    }
}
